// RSFM_52956946759771
// MI455X (gfx1250) — matrix operand order repaired and checked by reading, not compiled, not run on hardware
//
#include <hip/hip_runtime.h>
#include <hip/hip_bf16.h>

// ---------------------------------------------------------------------------
// Problem constants (match reference)
// ---------------------------------------------------------------------------
#define LSEQ 4096          // H*W
#define NB   2             // batch
#define DM   64            // d_model (= CIN)
#define DI   128           // d_inner
#define DSTATE 16          // d_state
#define DTR  32            // dt_rank
#define NROWS (NB * LSEQ)  // 8192 positions total
#define TILE_ROWS 64       // rows per workgroup (4 waves x 16-row WMMA tiles)

typedef __attribute__((ext_vector_type(16))) _Float16 v16h;
typedef __attribute__((ext_vector_type(8)))  _Float16 v8h;
typedef __attribute__((ext_vector_type(8)))  float    v8f;

// ---------------------------------------------------------------------------
// WMMA helpers (CDNA5: D = A(16x32 f16) x B(32x16 f16) + C(16x16 f32))
// ---------------------------------------------------------------------------
__device__ __forceinline__ v8f wmma32(v16h a, v16h b, v8f c) {
  return __builtin_amdgcn_wmma_f32_16x16x32_f16(
      /*neg_a=*/false, a, /*neg_b=*/false, b,
      /*c_mod=*/(short)0, c, /*reuse_a=*/false, /*reuse_b=*/false);
}

// A fragment from a row-major f16 tile in LDS (row stride ldk halves).
// Lanes 0-15: rows 0-15, K = k0+[0..15]
// Lanes 16-31: rows 0-15, K = k0+16+[0..15]
__device__ __forceinline__ v16h frag_a(const _Float16* tile, int ldk, int k0, int lane) {
  int m = lane & 15, grp = (lane >> 4) & 1;
  const _Float16* p = tile + m * ldk + k0 + grp * 16;
  v8h lo = *(const v8h*)p;
  v8h hi = *(const v8h*)(p + 8);
  return __builtin_shufflevector(lo, hi, 0,1,2,3,4,5,6,7,8,9,10,11,12,13,14,15);
}

// B fragment from LDS weights packed as [k/16][n][k%16] (contiguous 16 halves
// per (kblk,n)).  Lanes 0-15: K=k0+[0..15], col n0+lane; lanes 16-31: K=k0+16+[0..15].
__device__ __forceinline__ v16h frag_b(const _Float16* wp, int N, int k0, int n0, int lane) {
  int n = n0 + (lane & 15), grp = (lane >> 4) & 1;
  const _Float16* p = wp + (((k0 >> 4) + grp) * N + n) * 16;
  v8h lo = *(const v8h*)p;
  v8h hi = *(const v8h*)(p + 8);
  return __builtin_shufflevector(lo, hi, 0,1,2,3,4,5,6,7,8,9,10,11,12,13,14,15);
}

__device__ __forceinline__ float silu_f(float x) { return x / (1.f + __expf(-x)); }
__device__ __forceinline__ float softplus_f(float x) {
  return (x > 20.f) ? x : log1pf(__expf(x));
}

// ---------------------------------------------------------------------------
// 1x1 conv from [B, C, H*W] into row-major [B*L, O]  (conv1 / conv2)
// ---------------------------------------------------------------------------
__global__ void __launch_bounds__(256) k_proj_chw(
    const float* __restrict__ x, const float* __restrict__ w,
    const float* __restrict__ bias, float* __restrict__ out) {
  long long i = (long long)blockIdx.x * blockDim.x + threadIdx.x;
  if (i >= (long long)NROWS * DM) return;
  int o = (int)(i & (DM - 1));
  long long gr = i >> 6;                 // global row (b*LSEQ + l)
  int b = (int)(gr >> 12);
  int l = (int)(gr & (LSEQ - 1));
  const float* xp = x + ((long long)b * DM) * LSEQ + l;
  float acc = bias[o];
#pragma unroll 8
  for (int c = 0; c < DM; c++) acc += xp[(long long)c * LSEQ] * w[o * DM + c];
  out[gr * DM + o] = acc;
}

// 1x1 conv from row-major [B*L, C] into [B, O, H*W]  (conv3, final output)
__global__ void __launch_bounds__(256) k_proj_out(
    const float* __restrict__ x, const float* __restrict__ w,
    const float* __restrict__ bias, float* __restrict__ out) {
  long long i = (long long)blockIdx.x * blockDim.x + threadIdx.x;
  if (i >= (long long)NROWS * DM) return;
  int o = (int)(i & (DM - 1));
  long long gr = i >> 6;
  int b = (int)(gr >> 12);
  int l = (int)(gr & (LSEQ - 1));
  float acc = bias[o];
#pragma unroll 8
  for (int c = 0; c < DM; c++) acc += x[gr * DM + c] * w[o * DM + c];
  out[((long long)b * DM + o) * LSEQ + l] = acc;
}

// ---------------------------------------------------------------------------
// Kernel A: x = base (+prev); RMSNorm; in_proj GEMM [64 -> 256] via WMMA.
// Writes xn (f32, residual source), xc_raw (f32) and z (f32).
// Workgroup: 128 threads (4 waves), 64 rows.
// ---------------------------------------------------------------------------
__global__ void __launch_bounds__(128) k_norm_inproj(
    const float* __restrict__ base, const float* __restrict__ prev,
    const float* __restrict__ in_proj_w,  // [256, 64] row-major (o, c)
    const float* __restrict__ norm_w,
    float* __restrict__ xn_out, float* __restrict__ xc_raw,
    float* __restrict__ z_out) {
  __shared__ __align__(16) _Float16 s_xn[TILE_ROWS * DM];        // 8 KB
  __shared__ __align__(16) _Float16 s_w[(DM / 16) * 256 * 16];   // 32 KB
  __shared__ float s_rstd[TILE_ROWS];

  int tid = threadIdx.x, wid = tid >> 5, lane = tid & 31;
  long long row0 = (long long)blockIdx.x * TILE_ROWS;

  // Pack in_proj weights into B-fragment order: [k/16][n][k%16]
  for (int idx = tid; idx < DM * 256; idx += 128) {
    int n = idx >> 6, k = idx & 63;
    s_w[((k >> 4) * 256 + n) * 16 + (k & 15)] = (_Float16)in_proj_w[n * DM + k];
  }
  // Per-row rstd
  if (tid < TILE_ROWS) {
    long long gr = row0 + tid;
    float ss = 0.f;
#pragma unroll 8
    for (int c = 0; c < DM; c++) {
      float v = base[gr * DM + c] + (prev ? prev[gr * DM + c] : 0.f);
      ss += v * v;
    }
    s_rstd[tid] = rsqrtf(ss * (1.f / DM) + 1e-5f);
  }
  __syncthreads();

  // Normalize -> LDS f16 tile + global f32 (residual source)
  for (int idx = tid; idx < TILE_ROWS * DM; idx += 128) {
    int r = idx >> 6, c = idx & 63;
    long long gr = row0 + r;
    float v = base[gr * DM + c] + (prev ? prev[gr * DM + c] : 0.f);
    float xn = v * s_rstd[r] * norm_w[c];
    s_xn[r * DM + c] = (_Float16)xn;
    xn_out[gr * DM + c] = xn;
  }
  __syncthreads();

  // in_proj GEMM: each wave owns rows [wid*16, wid*16+16)
  const _Float16* at = s_xn + wid * 16 * DM;
  int grp = (lane >> 4) & 1, nlo = lane & 15;
  for (int n0 = 0; n0 < 256; n0 += 16) {
    v8f acc = {};
#pragma unroll
    for (int k0 = 0; k0 < DM; k0 += 32)
      acc = wmma32(frag_a(at, DM, k0, lane), frag_b(s_w, 256, k0, n0, lane), acc);
#pragma unroll
    for (int j = 0; j < 8; j++) {
      long long gr = row0 + wid * 16 + j + grp * 8;
      int n = n0 + nlo;
      if (n < DI) xc_raw[gr * DI + n] = acc[j];
      else        z_out[gr * DI + (n - DI)] = acc[j];
    }
  }
}

// ---------------------------------------------------------------------------
// Kernel B: dwconv1d+SiLU -> x_proj GEMM -> dt_proj GEMM (+bias, softplus)
//           -> per-position state update (h) -> gating -> out_proj GEMM (+xn)
// Workgroup: 128 threads (4 waves), 64 rows.  LDS ~76 KB.
// ---------------------------------------------------------------------------
__global__ void __launch_bounds__(128) k_scan(
    const float* __restrict__ xc_raw, const float* __restrict__ z_in,
    const float* __restrict__ xn_in,
    const float* __restrict__ conv1d_w,   // [128, 3]
    const float* __restrict__ conv1d_b,   // [128]
    const float* __restrict__ x_proj_w,   // [64, 128]
    const float* __restrict__ dt_proj_w,  // [128, 32]
    const float* __restrict__ dt_proj_b,  // [128]
    const float* __restrict__ A_log,      // [128, 16]
    const float* __restrict__ Dp,         // [128]
    const float* __restrict__ out_proj_w, // [64, 128]
    const float* __restrict__ h_in,       // [B*L, 128, 16] or null (zeros)
    float* __restrict__ h_out,
    float* __restrict__ out) {            // [B*L, 64]
  __shared__ __align__(16) _Float16 s_xc[TILE_ROWS * DI];       // 16 KB  conv+silu result
  __shared__ __align__(16) _Float16 s_wp[8 * DI * 16];          // 32 KB  shared weight pack
  __shared__ __align__(16) _Float16 s_dr[TILE_ROWS * DTR];      //  4 KB  delta_raw (A of dt GEMM)
  __shared__ float s_BC[TILE_ROWS * 32];                        //  8 KB  [r][0..15]=B, [16..31]=C
  __shared__ __align__(16) _Float16 s_dl[TILE_ROWS * DI];       // 16 KB  delta -> gated act

  int tid = threadIdx.x, wid = tid >> 5, lane = tid & 31;
  long long row0 = (long long)blockIdx.x * TILE_ROWS;  // tile never crosses batch bound
  int grp = (lane >> 4) & 1, nlo = lane & 15;

  // ---- Stage 1a: depthwise conv1d (k=3, pad=1 over L) + SiLU -> s_xc -------
  {
    int d = tid;  // 0..127, one channel per thread
    float cw0 = conv1d_w[d * 3 + 0], cw1 = conv1d_w[d * 3 + 1],
          cw2 = conv1d_w[d * 3 + 2], cb = conv1d_b[d];
    for (int r = 0; r < TILE_ROWS; r++) {
      long long gr = row0 + r;
      int l = (int)(gr & (LSEQ - 1));
      float acc = cb + cw1 * xc_raw[gr * DI + d];
      if (l > 0)        acc += cw0 * xc_raw[(gr - 1) * DI + d];
      if (l < LSEQ - 1) acc += cw2 * xc_raw[(gr + 1) * DI + d];
      s_xc[r * DI + d] = (_Float16)silu_f(acc);
    }
  }
  // ---- Stage 1b: pack x_proj weights (K=128, N=64) -------------------------
  for (int idx = tid; idx < 64 * DI; idx += 128) {
    int n = idx >> 7, k = idx & 127;
    s_wp[((k >> 4) * 64 + n) * 16 + (k & 15)] = (_Float16)x_proj_w[n * DI + k];
  }
  __syncthreads();

  // ---- Stage 2: x_proj GEMM -> delta_raw (f16) / Bm,Cm (f32) ---------------
  {
    const _Float16* at = s_xc + wid * 16 * DI;
    for (int n0 = 0; n0 < 64; n0 += 16) {
      v8f acc = {};
#pragma unroll
      for (int k0 = 0; k0 < DI; k0 += 32)
        acc = wmma32(frag_a(at, DI, k0, lane), frag_b(s_wp, 64, k0, n0, lane), acc);
#pragma unroll
      for (int j = 0; j < 8; j++) {
        int r = wid * 16 + j + grp * 8;
        int col = n0 + nlo;
        if (col < DTR) s_dr[r * DTR + col] = (_Float16)acc[j];
        else           s_BC[r * 32 + (col - DTR)] = acc[j];
      }
    }
  }
  __syncthreads();

  // ---- Stage 3a: pack dt_proj weights (K=32, N=128) ------------------------
  for (int idx = tid; idx < DI * DTR; idx += 128) {
    int n = idx >> 5, k = idx & 31;
    s_wp[((k >> 4) * DI + n) * 16 + (k & 15)] = (_Float16)dt_proj_w[n * DTR + k];
  }
  __syncthreads();

  // ---- Stage 3b: dt GEMM + bias + softplus -> s_dl (delta) ----------------
  {
    const _Float16* at = s_dr + wid * 16 * DTR;
    for (int n0 = 0; n0 < DI; n0 += 16) {
      v8f acc = {};
      acc = wmma32(frag_a(at, DTR, 0, lane), frag_b(s_wp, DI, 0, n0, lane), acc);
#pragma unroll
      for (int j = 0; j < 8; j++) {
        int r = wid * 16 + j + grp * 8;
        int col = n0 + nlo;
        s_dl[r * DI + col] = (_Float16)softplus_f(acc[j] + dt_proj_b[col]);
      }
    }
  }
  __syncthreads();

  // ---- Stage 4: per-position state update + gating (s_dl -> gated act) -----
  {
    int d = tid;
    float Aneg[DSTATE];
#pragma unroll
    for (int s = 0; s < DSTATE; s++) Aneg[s] = -__expf(A_log[d * DSTATE + s]);
    float Dpd = Dp[d];
    for (int r = 0; r < TILE_ROWS; r++) {
      long long gr = row0 + r;
      float delta = (float)s_dl[r * DI + d];
      float xcv = (float)s_xc[r * DI + d];
      float y = 0.f;
      long long hb = (gr * DI + d) * DSTATE;  // contiguous 16 floats per thread
#pragma unroll
      for (int s = 0; s < DSTATE; s++) {
        float hp = h_in ? h_in[hb + s] : 0.f;
        float hn = __expf(delta * Aneg[s]) * hp + delta * s_BC[r * 32 + s] * xcv;
        h_out[hb + s] = hn;
        y += hn * s_BC[r * 32 + 16 + s];
      }
      y += Dpd * xcv;
      float zv = z_in[gr * DI + d];
      s_dl[r * DI + d] = (_Float16)(y * silu_f(zv));  // same slot: no hazard
    }
  }
  // ---- Stage 5a: pack out_proj weights (K=128, N=64) — s_wp free now -------
  for (int idx = tid; idx < 64 * DI; idx += 128) {
    int n = idx >> 7, k = idx & 127;
    s_wp[((k >> 4) * 64 + n) * 16 + (k & 15)] = (_Float16)out_proj_w[n * DI + k];
  }
  __syncthreads();

  // ---- Stage 5b: out_proj GEMM + residual (xn) -> out ----------------------
  {
    const _Float16* at = s_dl + wid * 16 * DI;
    for (int n0 = 0; n0 < 64; n0 += 16) {
      v8f acc = {};
#pragma unroll
      for (int k0 = 0; k0 < DI; k0 += 32)
        acc = wmma32(frag_a(at, DI, k0, lane), frag_b(s_wp, 64, k0, n0, lane), acc);
#pragma unroll
      for (int j = 0; j < 8; j++) {
        long long gr = row0 + wid * 16 + j + grp * 8;
        int col = n0 + nlo;
        out[gr * DM + col] = acc[j] + xn_in[gr * DM + col];
      }
    }
  }
}

// ---------------------------------------------------------------------------
// Host launcher
// ---------------------------------------------------------------------------
extern "C" void kernel_launch(void* const* d_in, const int* in_sizes, int n_in,
                              void* d_out, int out_size, void* d_ws, size_t ws_size,
                              hipStream_t stream) {
  (void)in_sizes; (void)n_in; (void)out_size; (void)ws_size;
  const float* rgb       = (const float*)d_in[0];
  const float* dte       = (const float*)d_in[1];
  const float* conv1_w   = (const float*)d_in[2];
  const float* conv1_b   = (const float*)d_in[3];
  const float* conv2_w   = (const float*)d_in[4];
  const float* conv2_b   = (const float*)d_in[5];
  const float* conv3_w   = (const float*)d_in[6];
  const float* conv3_b   = (const float*)d_in[7];
  const float* norm_w    = (const float*)d_in[8];
  const float* in_proj_w = (const float*)d_in[9];
  const float* conv1d_w  = (const float*)d_in[10];
  const float* conv1d_b  = (const float*)d_in[11];
  const float* x_proj_w  = (const float*)d_in[12];
  const float* dt_proj_w = (const float*)d_in[13];
  const float* dt_proj_b = (const float*)d_in[14];
  const float* A_log     = (const float*)d_in[15];
  const float* Dp        = (const float*)d_in[16];
  const float* out_proj_w= (const float*)d_in[17];

  // Workspace (~16 MB): row-major [B*L, .] activation buffers
  float* ws       = (float*)d_ws;
  float* rgb_proj = ws;                                  // NROWS*DM
  float* dte_proj = rgb_proj + (long long)NROWS * DM;    // NROWS*DM
  float* xn       = dte_proj + (long long)NROWS * DM;    // NROWS*DM
  float* outb     = xn       + (long long)NROWS * DM;    // NROWS*DM
  float* xc_raw   = outb     + (long long)NROWS * DM;    // NROWS*DI
  float* zbuf     = xc_raw   + (long long)NROWS * DI;    // NROWS*DI

  // Output tuple: conv3 result [B,64,H,W] then h [B,L,128,16]; keep h in-place
  // in d_out across all 6 blocks (block 0 reads h as zeros via null pointer).
  float* conv_out = (float*)d_out;
  float* h_state  = conv_out + (long long)NROWS * DM;

  dim3 b256(256), gflat((NROWS * DM + 255) / 256);
  k_proj_chw<<<gflat, b256, 0, stream>>>(rgb, conv1_w, conv1_b, rgb_proj);
  k_proj_chw<<<gflat, b256, 0, stream>>>(dte, conv2_w, conv2_b, dte_proj);

  dim3 b128(128), gtile(NROWS / TILE_ROWS);
  for (int blk = 0; blk < 6; blk++) {
    const float* basep = (blk & 1) ? dte_proj : rgb_proj;
    const float* prevp = (blk == 0) ? nullptr : outb;
    k_norm_inproj<<<gtile, b128, 0, stream>>>(basep, prevp, in_proj_w, norm_w,
                                              xn, xc_raw, zbuf);
    const float* h_prev = (blk == 0) ? nullptr : h_state;
    k_scan<<<gtile, b128, 0, stream>>>(xc_raw, zbuf, xn, conv1d_w, conv1d_b,
                                       x_proj_w, dt_proj_w, dt_proj_b, A_log, Dp,
                                       out_proj_w, h_prev, h_state, outb);
  }
  k_proj_out<<<gflat, b256, 0, stream>>>(outb, conv3_w, conv3_b, conv_out);
}
